// MultiHeadedAttention_1967095021843
// MI455X (gfx1250) — matrix operand order repaired and checked by reading, not compiled, not run on hardware
//
#include <hip/hip_runtime.h>
#include <hip/hip_bf16.h>

// Rel-pos multi-head attention for MI455X (gfx1250), wave32 + WMMA f16,
// with async global->LDS staging (ASYNCcnt) in the large GEMMs.
//
// Operand conventions (CDNA5 ISA 7.12.2):
//   A (16x32 f16): lane L -> row m = L&15 ; g = L>>4 selects K subset
//                  {8g..8g+7, 16+8g..16+8g+7} -> two contiguous v8h loads.
//   B (32x16 f16): lane L -> col n = L&15 ; same K subset as A, strided by row.
//                  => B sources stored K-major (pre-transposed).
//   C/D (16x16 f32): element r -> m = r + 8*(L>>4), n = L&15.

typedef _Float16 half_t;
typedef __attribute__((ext_vector_type(16))) _Float16 v16h;
typedef __attribute__((ext_vector_type(8)))  _Float16 v8h;
typedef __attribute__((ext_vector_type(8)))  float    v8f;
typedef int i32x4 __attribute__((vector_size(16)));

#define BATCH 8
#define SEQ   1024
#define NF    1024
#define NH    16
#define DK    64
#define MINVAL (-3.4028234663852886e38f)

#if defined(__has_builtin)
#if __has_builtin(__builtin_amdgcn_global_load_async_to_lds_b128)
#define HAVE_ASYNC_LDS 1
#endif
#endif
#ifndef HAVE_ASYNC_LDS
#define HAVE_ASYNC_LDS 0
#endif

__device__ __forceinline__ void copy16_to_lds(const half_t* __restrict__ src,
                                              half_t* __restrict__ dst) {
#if HAVE_ASYNC_LDS
  // per-lane 16B async DMA into LDS, tracked by ASYNCcnt
  __builtin_amdgcn_global_load_async_to_lds_b128(
      (__attribute__((address_space(1))) i32x4*)src,
      (__attribute__((address_space(3))) i32x4*)dst, 0, 0);
#else
  *(uint4*)dst = *(const uint4*)src;
#endif
}

__device__ __forceinline__ void async_wait0() {
#if HAVE_ASYNC_LDS
#if __has_builtin(__builtin_amdgcn_s_wait_asynccnt)
  __builtin_amdgcn_s_wait_asynccnt(0);
#else
  asm volatile("s_wait_asynccnt 0" ::: "memory");
#endif
#endif
}

__device__ __forceinline__ v16h load_a16(const half_t* __restrict__ rowptr, int kb, int g) {
  const v8h lo = *(const v8h*)(rowptr + kb + 8 * g);
  const v8h hi = *(const v8h*)(rowptr + kb + 16 + 8 * g);
  v16h a;
#pragma unroll
  for (int i = 0; i < 8; ++i) { a[i] = lo[i]; a[i + 8] = hi[i]; }
  return a;
}

__device__ __forceinline__ v16h load_b16(const half_t* __restrict__ p, size_t stride, int g) {
  v16h b;
#pragma unroll
  for (int i = 0; i < 8; ++i) { b[i] = p[(size_t)(8 * g + i) * stride]; b[i + 8] = p[(size_t)(16 + 8 * g + i) * stride]; }
  return b;
}

__device__ __forceinline__ v8f wmma_f16(v16h a, v16h b, v8f c) {
  return __builtin_amdgcn_wmma_f32_16x16x32_f16(false, a, false, b, (short)0, c, false, false);
}

// ---------------- LayerNorm -> f16 ----------------
__global__ __launch_bounds__(256) void ln_kernel(const float* __restrict__ x,
                                                 const float* __restrict__ sc,
                                                 const float* __restrict__ bi,
                                                 half_t* __restrict__ xnh) {
  __shared__ float red[256], red2[256];
  const int row = blockIdx.x;
  const float* xr = x + (size_t)row * NF;
  float s = 0.f, s2 = 0.f;
  for (int i = threadIdx.x; i < NF; i += 256) { float v = xr[i]; s += v; s2 += v * v; }
  red[threadIdx.x] = s; red2[threadIdx.x] = s2;
  __syncthreads();
  for (int o = 128; o > 0; o >>= 1) {
    if ((int)threadIdx.x < o) { red[threadIdx.x] += red[threadIdx.x + o]; red2[threadIdx.x] += red2[threadIdx.x + o]; }
    __syncthreads();
  }
  const float mean = red[0] * (1.f / NF);
  const float var  = red2[0] * (1.f / NF) - mean * mean;
  const float rstd = rsqrtf(var + 1e-5f);
  half_t* orow = xnh + (size_t)row * NF;
  for (int i = threadIdx.x; i < NF; i += 256)
    orow[i] = (half_t)(((xr[i] - mean) * rstd) * sc[i] + bi[i]);
}

// ---------------- W (NxK f32) -> Wt (KxN f16) ----------------
__global__ __launch_bounds__(256) void wt_convert(const float* __restrict__ W,
                                                  half_t* __restrict__ Wt) {
  __shared__ float tile[32][33];
  const int j0 = blockIdx.x * 32, k0 = blockIdx.y * 32;
#pragma unroll
  for (int i = 0; i < 4; ++i) {
    int j = j0 + threadIdx.y + 8 * i;
    tile[threadIdx.y + 8 * i][threadIdx.x] = W[(size_t)j * NF + k0 + threadIdx.x];
  }
  __syncthreads();
#pragma unroll
  for (int i = 0; i < 4; ++i) {
    int k = k0 + threadIdx.y + 8 * i;
    Wt[(size_t)k * NF + j0 + threadIdx.x] = (half_t)tile[threadIdx.x][threadIdx.y + 8 * i];
  }
}

// ============== blocked GEMM: C = A(MxK) * Bt(KxN, K-major) + bias ================
// Block: 32 (M) x 256 (N); 8 waves in 2x4 grid; each wave: 16 x 64 (4 accums).
// B panel double-buffered in LDS via async global->LDS b128 copies.
#define BM 32
#define BN 256
#define BK 32
#define LDB 264  // halves per LDS panel row (528B, 16B-aligned, conflict-padded)

__device__ __forceinline__ void stage_panel(const half_t* __restrict__ Bt, int N,
                                            int kb, int nbase,
                                            half_t* __restrict__ panel) {
  // 32 rows x 256 halves = 1024 x 16B chunks; 4 per thread
  const int tid = threadIdx.x;
#pragma unroll
  for (int j = 0; j < 4; ++j) {
    const int c = tid + j * 256;
    const int row = c >> 5, cc = c & 31;
    copy16_to_lds(Bt + (size_t)(kb + row) * N + nbase + cc * 8,
                  panel + row * LDB + cc * 8);
  }
}

template <typename OutT>
__device__ __forceinline__ void gemm_body(const half_t* __restrict__ A,
                                          const half_t* __restrict__ Bt,
                                          const float* __restrict__ bias,
                                          OutT* __restrict__ C, int K, int N) {
  __shared__ half_t Bpan[2][BK * LDB];
  const int lane = threadIdx.x & 31, wave = threadIdx.x >> 5;
  const int g = lane >> 4, ln = lane & 15;
  const int wr = wave >> 2, wc = wave & 3;
  const int row0 = blockIdx.x * BM + wr * 16;
  const int nb = blockIdx.y * BN;
  const int n0 = wc * 64;
  const half_t* arow = A + (size_t)(row0 + ln) * K;
  v8f c0 = {}, c1 = {}, c2 = {}, c3 = {};

  stage_panel(Bt, N, 0, nb, Bpan[0]);
  int buf = 0;
  for (int kb = 0; kb < K; kb += BK, buf ^= 1) {
    async_wait0();
    __syncthreads();  // panel[buf] ready for all waves; prior reads of panel[buf^1] done
    if (kb + BK < K) stage_panel(Bt, N, kb + BK, nb, Bpan[buf ^ 1]);
    const v16h a = load_a16(arow, kb, g);
    const half_t* bl = &Bpan[buf][n0 + ln];
    c0 = wmma_f16(a, load_b16(bl, LDB, g), c0);
    c1 = wmma_f16(a, load_b16(bl + 16, LDB, g), c1);
    c2 = wmma_f16(a, load_b16(bl + 32, LDB, g), c2);
    c3 = wmma_f16(a, load_b16(bl + 48, LDB, g), c3);
  }

  const float b0 = bias[nb + n0 + ln];
  const float b1 = bias[nb + n0 + 16 + ln];
  const float b2 = bias[nb + n0 + 32 + ln];
  const float b3 = bias[nb + n0 + 48 + ln];
#pragma unroll
  for (int r = 0; r < 8; ++r) {
    const int m = r + 8 * g;
    OutT* crow = C + (size_t)(row0 + m) * N + nb + n0 + ln;
    crow[0]  = (OutT)(c0[r] + b0);
    crow[16] = (OutT)(c1[r] + b1);
    crow[32] = (OutT)(c2[r] + b2);
    crow[48] = (OutT)(c3[r] + b3);
  }
}

__global__ __launch_bounds__(256) void gemm_h(const half_t* __restrict__ A,
                                              const half_t* __restrict__ Bt,
                                              const float* __restrict__ bias,
                                              half_t* __restrict__ C, int K, int N) {
  gemm_body<half_t>(A, Bt, bias, C, K, N);
}

__global__ __launch_bounds__(256) void gemm_f(const half_t* __restrict__ A,
                                              const half_t* __restrict__ Bt,
                                              const float* __restrict__ bias,
                                              float* __restrict__ C, int K, int N) {
  gemm_body<float>(A, Bt, bias, C, K, N);
}

// ---------------- K (b,t,h,d) f16 -> Kt (b,h,d,t) f16 ----------------
__global__ __launch_bounds__(256) void kt_transpose(const half_t* __restrict__ Kh,
                                                    half_t* __restrict__ Kt) {
  __shared__ half_t tile[32][33];
  const int bh = blockIdx.x, b = bh >> 4, h = bh & 15;
  const int t0 = blockIdx.y * 32, d0 = blockIdx.z * 32;
#pragma unroll
  for (int i = 0; i < 4; ++i) {
    int t = t0 + threadIdx.y + 8 * i;
    tile[threadIdx.y + 8 * i][threadIdx.x] =
        Kh[((size_t)(b * SEQ + t) * NH + h) * DK + d0 + threadIdx.x];
  }
  __syncthreads();
#pragma unroll
  for (int i = 0; i < 4; ++i) {
    int d = d0 + threadIdx.y + 8 * i;
    Kt[((size_t)bh * DK + d) * SEQ + t0 + threadIdx.x] = tile[threadIdx.x][threadIdx.y + 8 * i];
  }
}

// ---------------- pos_k (t,s,d) f32 -> pkT (t,d,s) f16 (L2-resident, 128MB) --------
__global__ __launch_bounds__(256) void pk_transpose(const float* __restrict__ pk,
                                                    half_t* __restrict__ pkT) {
  __shared__ float tile[32][33];
  const int t = blockIdx.x, s0 = blockIdx.y * 32, d0 = blockIdx.z * 32;
#pragma unroll
  for (int i = 0; i < 4; ++i) {
    int s = s0 + threadIdx.y + 8 * i;
    tile[threadIdx.y + 8 * i][threadIdx.x] = pk[((size_t)t * SEQ + s) * DK + d0 + threadIdx.x];
  }
  __syncthreads();
#pragma unroll
  for (int i = 0; i < 4; ++i) {
    int d = d0 + threadIdx.y + 8 * i;
    pkT[((size_t)t * DK + d) * SEQ + s0 + threadIdx.x] =
        (half_t)tile[threadIdx.x][threadIdx.y + 8 * i];
  }
}

// ---------------- B-score: for fixed t, Bt = Q_bh(128x64) x pkT[t](64x1024) -------
__global__ __launch_bounds__(256) void bscore_kernel(const half_t* __restrict__ Qh,
                                                     const half_t* __restrict__ pkT,
                                                     half_t* __restrict__ Bsc) {
  const int lane = threadIdx.x & 31, wave = threadIdx.x >> 5;
  const int g = lane >> 4, ln = lane & 15;
  const int t = blockIdx.x;
  const int s0 = blockIdx.y * 16;
  const int bh = wave * 16 + ln;          // A row index
  const int b = bh >> 4, h = bh & 15;
  const half_t* qrow = Qh + ((size_t)(b * SEQ + t) * NH + h) * DK;
  v8f c = {};
#pragma unroll
  for (int kb = 0; kb < DK; kb += 32) {
    v16h a = load_a16(qrow, kb, g);
    v16h bb = load_b16(pkT + ((size_t)t * DK + kb) * SEQ + s0 + ln, SEQ, g);
    c = wmma_f16(a, bb, c);
  }
#pragma unroll
  for (int r = 0; r < 8; ++r) {
    int mbh = wave * 16 + r + 8 * g;
    Bsc[((size_t)mbh * SEQ + t) * SEQ + s0 + ln] = (half_t)c[r];
  }
}

// ---------------- fused scores + softmax + attn@V (one (b,h,t-tile) per WG) -------
#define SROW 1028   // f32 score row stride (pad -> conflict-free)
#define AROW 1032   // f16 attn row stride

__global__ __launch_bounds__(256) void attn_kernel(const half_t* __restrict__ Qh,
                                                   const half_t* __restrict__ Kt,
                                                   const half_t* __restrict__ Vh,
                                                   const half_t* __restrict__ Bsc,
                                                   const int* __restrict__ mask,
                                                   half_t* __restrict__ AO) {
  extern __shared__ char smem[];
  float*  sc = (float*)smem;                                   // 16 x SROW f32
  half_t* at = (half_t*)(smem + 16 * SROW * sizeof(float));    // 16 x AROW f16
  const int lane = threadIdx.x & 31, wave = threadIdx.x >> 5;
  const int g = lane >> 4, ln = lane & 15;
  const int b = blockIdx.z, h = blockIdx.y;
  const int t0 = blockIdx.x * 16;
  const int bh = b * NH + h;

  // Q tile A-operands (d = 0..63 -> two K=32 chunks), fixed for this WG
  const half_t* qrow = Qh + ((size_t)(b * SEQ + t0 + ln) * NH + h) * DK;
  const v16h a0 = load_a16(qrow, 0, g);
  const v16h a1 = load_a16(qrow, 32, g);

  // ---- phase 1: scores = (Bsc + Q K^T) / 8, masked; 64 s-tiles over 8 waves ----
  for (int st = wave; st < SEQ / 16; st += 8) {
    const int s0 = st * 16;
    v8f c = {};
#pragma unroll
    for (int r = 0; r < 8; ++r) {   // seed accumulator with rel-pos score tile
      int m = r + 8 * g;
      c[r] = (float)Bsc[((size_t)bh * SEQ + t0 + m) * SEQ + s0 + ln];
    }
    c = wmma_f16(a0, load_b16(Kt + ((size_t)bh * DK) * SEQ + s0 + ln, SEQ, g), c);
    c = wmma_f16(a1, load_b16(Kt + ((size_t)bh * DK + 32) * SEQ + s0 + ln, SEQ, g), c);
#pragma unroll
    for (int r = 0; r < 8; ++r) {
      int m = r + 8 * g;
      float v = c[r] * 0.125f;  // 1/sqrt(64)
      if (mask[((size_t)b * SEQ + t0 + m) * SEQ + s0 + ln] == 0) v = MINVAL;
      sc[m * SROW + s0 + ln] = v;
    }
  }
  __syncthreads();

  // ---- phase 2: exact softmax, one wave per 2 rows ----
  for (int rr = 0; rr < 2; ++rr) {
    const int row = wave * 2 + rr;
    float mx = MINVAL;
    for (int i = lane; i < SEQ; i += 32) mx = fmaxf(mx, sc[row * SROW + i]);
#pragma unroll
    for (int o = 16; o > 0; o >>= 1) mx = fmaxf(mx, __shfl_xor(mx, o, 32));
    float sum = 0.f;
    for (int i = lane; i < SEQ; i += 32) {
      float v = sc[row * SROW + i];
      float e = (v <= -1e37f) ? 0.f : __expf(v - mx);  // masked -> exactly 0
      sc[row * SROW + i] = e;
      sum += e;
    }
#pragma unroll
    for (int o = 16; o > 0; o >>= 1) sum += __shfl_xor(sum, o, 32);
    const float inv = (sum > 0.f) ? 1.f / sum : 0.f;   // fully-masked row -> zeros
    for (int i = lane; i < SEQ; i += 32)
      at[row * AROW + i] = (half_t)(sc[row * SROW + i] * inv);
  }
  __syncthreads();

  // ---- phase 3: attn @ V ; V(b,s,h,d) is already [K=s][N=d]-major ----
  if (wave < 4) {
    const int d0 = wave * 16;
    const half_t* arow = at + (size_t)ln * AROW;
    v8f c = {};
    for (int sb = 0; sb < SEQ; sb += 32) {
      v16h a = load_a16(arow, sb, g);
      v16h bb = load_b16(Vh + ((size_t)(b * SEQ + sb) * NH + h) * DK + d0 + ln, (size_t)NH * DK, g);
      c = wmma_f16(a, bb, c);
    }
#pragma unroll
    for (int r = 0; r < 8; ++r) {
      int m = r + 8 * g;
      AO[(size_t)(b * SEQ + t0 + m) * NF + h * DK + d0 + ln] = (half_t)c[r];
    }
  }
}

// ---------------------------------------------------------------------------------
extern "C" void kernel_launch(void* const* d_in, const int* in_sizes, int n_in,
                              void* d_out, int out_size, void* d_ws, size_t ws_size,
                              hipStream_t stream) {
  const float* x        = (const float*)d_in[0];
  const float* pos_k    = (const float*)d_in[1];
  const int*   mask     = (const int*)d_in[2];
  const float* ln_scale = (const float*)d_in[3];
  const float* ln_bias  = (const float*)d_in[4];
  const float* wq = (const float*)d_in[5];  const float* bq = (const float*)d_in[6];
  const float* wk = (const float*)d_in[7];  const float* bk = (const float*)d_in[8];
  const float* wv = (const float*)d_in[9];  const float* bv = (const float*)d_in[10];
  const float* wo = (const float*)d_in[11]; const float* bo = (const float*)d_in[12];
  float* out = (float*)d_out;

  // workspace carve-up (~489 MB total)
  char* ws = (char*)d_ws;
  size_t off = 0;
  auto alloc = [&](size_t bytes) -> char* {
    char* p = ws + off;
    off += (bytes + 255) & ~(size_t)255;
    return p;
  };
  const size_t ROWS = (size_t)BATCH * SEQ;                 // 8192
  half_t* xnh = (half_t*)alloc(ROWS * NF * 2);             // 16 MB
  half_t* wqt = (half_t*)alloc((size_t)NF * NF * 2);       // 2 MB
  half_t* wkt = (half_t*)alloc((size_t)NF * NF * 2);
  half_t* wvt = (half_t*)alloc((size_t)NF * NF * 2);
  half_t* wot = (half_t*)alloc((size_t)NF * NF * 2);
  half_t* Qh  = (half_t*)alloc(ROWS * NF * 2);             // 16 MB (b,t,h,d)
  half_t* Kh  = (half_t*)alloc(ROWS * NF * 2);             // 16 MB (b,t,h,d)
  half_t* Vh  = (half_t*)alloc(ROWS * NF * 2);             // 16 MB (b,s,h,d)
  half_t* Kt  = (half_t*)alloc(ROWS * NF * 2);             // 16 MB (b,h,d,t)
  half_t* pkT = (half_t*)alloc((size_t)SEQ * DK * SEQ * 2);        // 128 MB (t,d,s)
  half_t* Bsc = (half_t*)alloc((size_t)BATCH * NH * SEQ * SEQ * 2);// 256 MB (bh,t,s)
  half_t* AO  = (half_t*)alloc(ROWS * NF * 2);             // 16 MB (b,t,f)

  // 1. LayerNorm -> f16
  ln_kernel<<<ROWS, 256, 0, stream>>>(x, ln_scale, ln_bias, xnh);

  // 2. weight transpose+convert (K-major for WMMA B-operand)
  wt_convert<<<dim3(32, 32), dim3(32, 8), 0, stream>>>(wq, wqt);
  wt_convert<<<dim3(32, 32), dim3(32, 8), 0, stream>>>(wk, wkt);
  wt_convert<<<dim3(32, 32), dim3(32, 8), 0, stream>>>(wv, wvt);
  wt_convert<<<dim3(32, 32), dim3(32, 8), 0, stream>>>(wo, wot);

  // 3. QKV projections (WMMA, async-LDS double-buffered B panels)
  gemm_h<<<dim3(ROWS / BM, NF / BN), 256, 0, stream>>>(xnh, wqt, bq, Qh, NF, NF);
  gemm_h<<<dim3(ROWS / BM, NF / BN), 256, 0, stream>>>(xnh, wkt, bk, Kh, NF, NF);
  gemm_h<<<dim3(ROWS / BM, NF / BN), 256, 0, stream>>>(xnh, wvt, bv, Vh, NF, NF);

  // 4. K -> Kt (d-major), pos_k -> pkT f16 (d-major; 128MB -> L2-resident)
  kt_transpose<<<dim3(BATCH * NH, SEQ / 32, DK / 32), dim3(32, 8), 0, stream>>>(Kh, Kt);
  pk_transpose<<<dim3(SEQ, SEQ / 32, DK / 32), dim3(32, 8), 0, stream>>>(pos_k, pkT);

  // 5. rel-pos scores via per-t GEMM (WMMA): Bsc[bh,t,s]
  bscore_kernel<<<dim3(SEQ, SEQ / 16), 256, 0, stream>>>(Qh, pkT, Bsc);

  // 6. fused scores + softmax + attn@V (WMMA), 99KB dynamic LDS
  const size_t smem = 16 * SROW * sizeof(float) + 16 * AROW * sizeof(half_t);
  attn_kernel<<<dim3(SEQ / 16, NH, BATCH), 256, smem, stream>>>(Qh, Kt, Vh, Bsc, mask, AO);

  // 7. output projection (WMMA, f32 out)
  gemm_f<<<dim3(ROWS / BM, NF / BN), 256, 0, stream>>>(AO, wot, bo, out, NF, NF);
}
